// EdgeAwareSpatialAttention_69818988364299
// MI455X (gfx1250) — matrix operand order repaired and checked by reading, not compiled, not run on hardware
//
#include <hip/hip_runtime.h>

// ---------------------------------------------------------------------------
// EdgeAwareSpatialAttention for MI455X (gfx1250, wave32, WMMA bf16,
// async global->LDS staging). B=8, N=1024, D=256, H=8, Hd=32
// ---------------------------------------------------------------------------

typedef __attribute__((ext_vector_type(16))) __bf16 v16bf;
typedef __attribute__((ext_vector_type(8)))  __bf16 v8bf;
typedef __attribute__((ext_vector_type(8)))  float  v8f;

#define BATCH  8
#define NTOT   1024
#define DDIM   256
#define NHEAD  8
#define HDIM   32
#define ROWS   (BATCH * NTOT)          // 8192 flattened (b,n) rows

// ---- WMMA helpers ----------------------------------------------------------

__device__ __forceinline__ v8f wmma_bf16(v16bf a, v16bf b, v8f c) {
  // D = A(16x32 bf16) * B(32x16 bf16) + C(16x16 f32)
  return __builtin_amdgcn_wmma_f32_16x16x32_bf16(false, a, false, b,
                                                 (short)0, c, false, false);
}

// A fragment (16x32, bf16) from a row-major tile; caller passes this lane's
// row pointer (row m = lane&15).
// ISA layout: half i -> K = (i&8)*2 + hi*8 + (i&7)  => two contiguous 16B runs.
__device__ __forceinline__ v16bf a_frag_bf16(const __bf16* row, int hi) {
  union { v16bf v; v8bf h[2]; } u;
  u.h[0] = *reinterpret_cast<const v8bf*>(row + hi * 8);
  u.h[1] = *reinterpret_cast<const v8bf*>(row + 16 + hi * 8);
  return u.v;
}

// B fragment (32x16, bf16) from storage laid out [n][K] row-major.
// ISA layout: lane n = lane&15, half i -> K = (i&8)*2 + hi*8 + (i&7) => two 16B runs.
__device__ __forceinline__ v16bf b_frag(const __bf16* base, int ld, int lane) {
  int n = lane & 15, hi = (lane >> 4) & 1;
  const __bf16* p = base + (size_t)n * ld + hi * 8;
  union { v16bf v; v8bf h[2]; } u;
  u.h[0] = *reinterpret_cast<const v8bf*>(p);
  u.h[1] = *reinterpret_cast<const v8bf*>(p + 16);
  return u.v;
}

__device__ __forceinline__ v8f zero8() {
  v8f z;
#pragma unroll
  for (int r = 0; r < 8; ++r) z[r] = 0.f;
  return z;
}

// ---- Kernel 0: xw[b,n] = X[b,n,:] . w_adj ---------------------------------

__global__ __launch_bounds__(256)
void xw_kernel(const float* __restrict__ X, const float* __restrict__ wadj,
               float* __restrict__ xw) {
  int w = threadIdx.x >> 5, lane = threadIdx.x & 31;
  int row = blockIdx.x * 8 + w;            // 1024 blocks * 8 waves = 8192 rows
  const float* xr = X + (size_t)row * DDIM;
  float s = 0.f;
#pragma unroll
  for (int k = 0; k < 8; ++k) s += xr[lane + 32 * k] * wadj[lane + 32 * k];
#pragma unroll
  for (int m = 16; m > 0; m >>= 1) s += __shfl_xor(s, m, 32);
  if (lane == 0) xw[row] = s;
}

// ---- Kernel 1: stream A once; write adj_update and per-head modulator -----

__global__ __launch_bounds__(256)
void adj_kernel(const float* __restrict__ A, const float* __restrict__ xw,
                const float* __restrict__ Wa, const float* __restrict__ ba,
                const float* __restrict__ badj,
                float* __restrict__ adjo, float* __restrict__ mod) {
  __shared__ float red[256];
  __shared__ float smwa[NHEAD], smba[NHEAD];
  int t = threadIdx.x;
  int row = blockIdx.x;                    // (b,i) flattened: 8192 blocks
  int b = row >> 10, i = row & 1023;

  if (t < NHEAD) {                         // per-head means of Wa / ba
    float wa = 0.f, bb = 0.f;
    for (int d = 0; d < HDIM; ++d) { wa += Wa[t * HDIM + d]; bb += ba[t * HDIM + d]; }
    smwa[t] = wa * (1.f / HDIM);
    smba[t] = bb * (1.f / HDIM);
  }

  float bav = badj[0];
  const float* arow = A + (size_t)row * NTOT;
  float*       orow = adjo + (size_t)row * NTOT;
  const float* xwb  = xw + (size_t)b * NTOT;

  float s = 0.f;
#pragma unroll
  for (int j = t; j < NTOT; j += 256) {    // fused: row sum + adj_update write
    float a = arow[j];
    s += a;
    orow[j] = fmaf(a, xwb[j], bav);
  }
  red[t] = s;
  __syncthreads();
  for (int off = 128; off > 0; off >>= 1) {
    if (t < off) red[t] += red[t + off];
    __syncthreads();
  }
  float amean = red[0] * (1.f / NTOT);
  if (t < NHEAD)
    mod[((size_t)b * NHEAD + t) * NTOT + i] = fmaf(amean, smwa[t], smba[t]);
}

// ---- Kernel 2: QKV projection (X @ W + b), WMMA, bf16 outputs --------------
// blockIdx.z: 0=Q, 1=K (layout (B,H,N,Hd)), 2=V transposed (B,H,Hd,N)

__global__ __launch_bounds__(256)
void qkv_kernel(const float* __restrict__ X,
                const float* __restrict__ Wq, const float* __restrict__ bq,
                const float* __restrict__ Wk, const float* __restrict__ bk_,
                const float* __restrict__ Wv, const float* __restrict__ bv_,
                __bf16* __restrict__ qo, __bf16* __restrict__ ko,
                __bf16* __restrict__ vo) {
  __shared__ __attribute__((aligned(16))) __bf16 Wt[16][40];  // [n][k], padded
  const int tid = threadIdx.x, w = tid >> 5, lane = tid & 31;
  const int n16 = lane & 15, hi = (lane >> 4) & 1;
  const int m0 = blockIdx.x * 128 + w * 16;   // 64 blocks * 8 waves * 16 rows
  const int n0 = blockIdx.y * 16;             // 16 col tiles
  const int mode = blockIdx.z;

  const float* W    = (mode == 0) ? Wq : (mode == 1) ? Wk : Wv;
  const float* bias = (mode == 0) ? bq : (mode == 1) ? bk_ : bv_;

  v8f acc = zero8();
  for (int k = 0; k < DDIM; k += 32) {
    __syncthreads();
    // stage W tile 32(k) x 16(n), transposed to [n][k] bf16 in LDS
    for (int e = tid; e < 512; e += 256) {
      int kk = e >> 4, nn = e & 15;
      Wt[nn][kk] = (__bf16)W[(size_t)(k + kk) * DDIM + n0 + nn];
    }
    __syncthreads();

    // A fragment straight from X with f32->bf16 converts
    const float* xr = X + (size_t)(m0 + n16) * DDIM + k;
    v16bf a;
#pragma unroll
    for (int i2 = 0; i2 < 8; ++i2) {
      a[i2]     = (__bf16)xr[hi * 8 + i2];
      a[8 + i2] = (__bf16)xr[16 + hi * 8 + i2];
    }
    v16bf bfr = b_frag(&Wt[0][0], 40, lane);
    acc = wmma_bf16(a, bfr, acc);
  }

  float bv = bias[n0 + n16];
#pragma unroll
  for (int r = 0; r < 8; ++r) {
    float v  = acc[r] + bv;                  // C layout: m = r + hi*8, n = lane&15
    int rowg = m0 + r + hi * 8;
    int b    = rowg >> 10, n = rowg & 1023;
    int col  = n0 + n16;
    int h    = col >> 5, hd = col & 31;
    __bf16 val = (__bf16)v;
    if (mode == 2)
      vo[(((size_t)b * NHEAD + h) * HDIM + hd) * NTOT + n] = val;   // (B,H,Hd,N)
    else {
      __bf16* dst = (mode == 0) ? qo : ko;
      dst[(((size_t)b * NHEAD + h) * NTOT + n) * HDIM + hd] = val;  // (B,H,N,Hd)
    }
  }
}

// ---- Kernel 3: modulated attention, flash-style online softmax -------------
// grid: 512 blocks; block = 8 waves; each wave owns 16 query rows of one (b,h).
// K (1024x32) and V^T (32x1024) for the block's head are staged in LDS via
// the async global->LDS engine (shared by all 8 waves: 8x less HBM traffic).

__global__ __launch_bounds__(256)
void attn_kernel(const __bf16* __restrict__ qb, const __bf16* __restrict__ kb,
                 const __bf16* __restrict__ vt, const float* __restrict__ mod,
                 __bf16* __restrict__ ao) {
  __shared__ __attribute__((aligned(16))) __bf16 Ks[NTOT][40];       // 80KB [j][hd] padded
  __shared__ __attribute__((aligned(16))) __bf16 Vs[HDIM][NTOT + 8]; // 66KB [hd][j] padded
  __shared__ __attribute__((aligned(16))) __bf16 plds[8][16][40];    // 10KB P staging

  const int tid = threadIdx.x, w = tid >> 5, lane = tid & 31;
  const int n16 = lane & 15, hi = (lane >> 4) & 1;
  const int bh = blockIdx.x >> 3;                 // (b*8+h)
  const int q0 = (blockIdx.x & 7) * 128 + w * 16; // query tile start
  const int b = bh >> 3, h = bh & 7;
  const float scaling = 0.1767766952966369f;      // 1/sqrt(32)

  // ---- async-stage K and V^T for this (b,h): 2 x 64KB, 16B chunks ---------
  {
    const __bf16* kg = kb + (size_t)bh * NTOT * HDIM;
    const __bf16* vg = vt + (size_t)bh * HDIM * NTOT;
    unsigned kls = (unsigned)(uintptr_t)&Ks[0][0];  // LDS byte offset (addr[31:0])
    unsigned vls = (unsigned)(uintptr_t)&Vs[0][0];
#pragma unroll
    for (int t = 0; t < 16; ++t) {
      int q = tid + t * 256;                        // 16B chunk id, 0..4095
      // K: global row j = 64B  ->  LDS row j = 80B (padded)
      unsigned kgo = (unsigned)((q >> 2) * 64 + (q & 3) * 16);
      unsigned klo = kls + (unsigned)((q >> 2) * 80 + (q & 3) * 16);
      asm volatile("global_load_async_to_lds_b128 %0, %1, %2"
                   :: "v"(klo), "v"(kgo), "s"(kg) : "memory");
      // V: global row hd = 2048B -> LDS row hd = 2064B (padded)
      unsigned vgo = (unsigned)((q >> 7) * 2048 + (q & 127) * 16);
      unsigned vlo = vls + (unsigned)((q >> 7) * 2064 + (q & 127) * 16);
      asm volatile("global_load_async_to_lds_b128 %0, %1, %2"
                   :: "v"(vlo), "v"(vgo), "s"(vg) : "memory");
    }
    asm volatile("s_wait_asynccnt 0" ::: "memory");
  }
  __syncthreads();

  // Q A-fragment: loaded once, reused for all 64 key tiles
  const __bf16* qrow = qb + ((size_t)bh * NTOT + q0 + n16) * HDIM;
  v16bf qa = a_frag_bf16(qrow, hi);

  float modrow[8], rmax[8], rsum[8];
#pragma unroll
  for (int r = 0; r < 8; ++r) {
    modrow[r] = mod[(size_t)bh * NTOT + q0 + r + hi * 8] * scaling;
    rmax[r] = -3.0e38f;
    rsum[r] = 0.f;
  }
  v8f acc0 = zero8(), acc1 = zero8();

  __bf16(*pw)[40] = plds[w];

  for (int js = 0; js < NTOT; js += 32) {
    // scores for two 16-wide key tiles (Hd=32 == one WMMA K-depth)
    v16bf bk0 = b_frag(&Ks[js][0], 40, lane);
    v16bf bk1 = b_frag(&Ks[js + 16][0], 40, lane);
    v8f s0 = wmma_bf16(qa, bk0, zero8());
    v8f s1 = wmma_bf16(qa, bk1, zero8());

#pragma unroll
    for (int r = 0; r < 8; ++r) {
      float v0 = s0[r] * modrow[r];
      float v1 = s1[r] * modrow[r];
      // row max over 16 lanes of this half (lanes 0-15 / 16-31 independent)
      float tmax = fmaxf(v0, v1);
      tmax = fmaxf(tmax, __shfl_xor(tmax, 1, 32));
      tmax = fmaxf(tmax, __shfl_xor(tmax, 2, 32));
      tmax = fmaxf(tmax, __shfl_xor(tmax, 4, 32));
      tmax = fmaxf(tmax, __shfl_xor(tmax, 8, 32));
      float nm   = fmaxf(rmax[r], tmax);
      float corr = __expf(rmax[r] - nm);
      rmax[r] = nm;
      float p0 = __expf(v0 - nm);
      float p1 = __expf(v1 - nm);
      float ps = p0 + p1;
      ps += __shfl_xor(ps, 1, 32);
      ps += __shfl_xor(ps, 2, 32);
      ps += __shfl_xor(ps, 4, 32);
      ps += __shfl_xor(ps, 8, 32);
      rsum[r] = rsum[r] * corr + ps;
      acc0[r] *= corr;
      acc1[r] *= corr;
      int m = r + hi * 8;                      // C-layout -> LDS [m][j] tile
      pw[m][n16]      = (__bf16)p0;
      pw[m][16 + n16] = (__bf16)p1;
    }
    asm volatile("s_wait_dscnt 0" ::: "memory"); // cross-lane LDS visibility

    // P (16x32) as A-fragment, V tiles as B-fragments; accumulate O
    v16bf pa  = a_frag_bf16(&pw[n16][0], hi);
    v16bf bv0 = b_frag(&Vs[0][js], NTOT + 8, lane);
    v16bf bv1 = b_frag(&Vs[16][js], NTOT + 8, lane);
    acc0 = wmma_bf16(pa, bv0, acc0);
    acc1 = wmma_bf16(pa, bv1, acc1);
  }

  // epilogue: divide by softmax denom, store (B,N,D) bf16
#pragma unroll
  for (int r = 0; r < 8; ++r) {
    int m = r + hi * 8;
    int n = q0 + m;
    float inv = 1.f / rsum[r];
    size_t base = ((size_t)b * NTOT + n) * DDIM + h * HDIM;
    ao[base + n16]      = (__bf16)(acc0[r] * inv);
    ao[base + 16 + n16] = (__bf16)(acc1[r] * inv);
  }
}

// ---- Kernel 4: output projection (attn_out @ Wo + bo) -> f32 ---------------

__global__ __launch_bounds__(256)
void oproj_kernel(const __bf16* __restrict__ ao, const float* __restrict__ Wo,
                  const float* __restrict__ bo, float* __restrict__ out) {
  __shared__ __attribute__((aligned(16))) __bf16 Wt[16][40];
  const int tid = threadIdx.x, w = tid >> 5, lane = tid & 31;
  const int n16 = lane & 15, hi = (lane >> 4) & 1;
  const int m0 = blockIdx.x * 128 + w * 16;
  const int n0 = blockIdx.y * 16;

  v8f acc = zero8();
  for (int k = 0; k < DDIM; k += 32) {
    __syncthreads();
    for (int e = tid; e < 512; e += 256) {
      int kk = e >> 4, nn = e & 15;
      Wt[nn][kk] = (__bf16)Wo[(size_t)(k + kk) * DDIM + n0 + nn];
    }
    __syncthreads();
    const __bf16* ar = ao + (size_t)(m0 + n16) * DDIM + k;
    v16bf a   = a_frag_bf16(ar, hi);
    v16bf bfr = b_frag(&Wt[0][0], 40, lane);
    acc = wmma_bf16(a, bfr, acc);
  }
  float bv = bo[n0 + n16];
#pragma unroll
  for (int r = 0; r < 8; ++r) {
    int rowg = m0 + r + hi * 8;
    out[(size_t)rowg * DDIM + n0 + n16] = acc[r] + bv;
  }
}

// ---------------------------------------------------------------------------

extern "C" void kernel_launch(void* const* d_in, const int* in_sizes, int n_in,
                              void* d_out, int out_size, void* d_ws,
                              size_t ws_size, hipStream_t stream) {
  (void)in_sizes; (void)n_in; (void)out_size; (void)ws_size;
  const float* X    = (const float*)d_in[0];
  const float* A    = (const float*)d_in[1];
  const float* Wq   = (const float*)d_in[2];
  const float* bq   = (const float*)d_in[3];
  const float* Wk   = (const float*)d_in[4];
  const float* bk   = (const float*)d_in[5];
  const float* Wv   = (const float*)d_in[6];
  const float* bv   = (const float*)d_in[7];
  const float* Wa   = (const float*)d_in[8];
  const float* ba   = (const float*)d_in[9];
  const float* Wo   = (const float*)d_in[10];
  const float* bo   = (const float*)d_in[11];
  const float* wadj = (const float*)d_in[12];
  const float* badj = (const float*)d_in[13];

  float* out_feat = (float*)d_out;                         // (B,N,D)
  float* out_adj  = out_feat + (size_t)ROWS * DDIM;        // (B,N,N)

  char* ws = (char*)d_ws;
  __bf16* qb  = (__bf16*)(ws);                             // (B,H,N,Hd)  4MB
  __bf16* kb  = (__bf16*)(ws + (size_t)4  * 1024 * 1024);  // (B,H,N,Hd)  4MB
  __bf16* vt  = (__bf16*)(ws + (size_t)8  * 1024 * 1024);  // (B,H,Hd,N)  4MB
  __bf16* ao  = (__bf16*)(ws + (size_t)12 * 1024 * 1024);  // (B,N,D)     4MB
  float*  xw  = (float*)(ws + (size_t)16 * 1024 * 1024);   // (B,N)      32KB
  float*  mod = (float*)(ws + (size_t)16 * 1024 * 1024 + 64 * 1024); // (B,H,N)

  xw_kernel<<<dim3(ROWS / 8), dim3(256), 0, stream>>>(X, wadj, xw);
  adj_kernel<<<dim3(ROWS), dim3(256), 0, stream>>>(A, xw, Wa, ba, badj,
                                                   out_adj, mod);
  qkv_kernel<<<dim3(ROWS / 128, DDIM / 16, 3), dim3(256), 0, stream>>>(
      X, Wq, bq, Wk, bk, Wv, bv, qb, kb, vt);
  attn_kernel<<<dim3(BATCH * NHEAD * 8), dim3(256), 0, stream>>>(qb, kb, vt,
                                                                 mod, ao);
  oproj_kernel<<<dim3(ROWS / 128, DDIM / 16), dim3(256), 0, stream>>>(
      ao, Wo, bo, out_feat);
}
